// GraphConv_71975061946781
// MI455X (gfx1250) — compile-verified, matrix operand order checked by reading, not run on hardware
//
#include <hip/hip_runtime.h>
#include <hip/hip_bf16.h>
#include <math.h>

// ---------------------------------------------------------------------------
// PaiNN-style GNN layer for MI455X (gfx1250, wave32, WMMA).
//   - edge MLP2 (25.2 GFLOP, dominant)  -> v_wmma_f32_16x16x32_f16
//   - node pre/post GEMMs (accuracy-critical) -> v_wmma_f32_16x16x4_f32
//   - edge_attr staging -> global_load_async_to_lds_b64 (ASYNCcnt path)
//   - segment_sum -> global fp32 atomics into L2-resident accumulators
// ---------------------------------------------------------------------------

typedef _Float16 v16h __attribute__((ext_vector_type(16)));
typedef _Float16 v8h  __attribute__((ext_vector_type(8)));
typedef float    v8f  __attribute__((ext_vector_type(8)));
typedef float    v2f  __attribute__((ext_vector_type(2)));

#define G_NS      128
#define G_NV      64
#define G_TPC     192
#define G_NODES   16000
#define G_EDGES   256000

__device__ __forceinline__ float fsigmoid(float x) { return 1.0f / (1.0f + __expf(-x)); }

// one K=4 step of the fp32 WMMA (A 16x4, B 4x16, C/D 16x16 f32)
__device__ __forceinline__ v8f wmma_f32x4(float a0, float a1, float b0, float b1, v8f c) {
    v2f a, b;
    a[0] = a0; a[1] = a1;
    b[0] = b0; b[1] = b1;
    return __builtin_amdgcn_wmma_f32_16x16x4_f32(false, a, false, b, (short)0, c, false, false);
}

// ---------------------------------------------------------------------------
// K0: W2 (128x384 f32, row-major [K][N]) -> W2ht (384x128 f16, [N][K])
// ---------------------------------------------------------------------------
__global__ void convert_w2_kernel(const float* __restrict__ W2, _Float16* __restrict__ W2ht) {
    int i = blockIdx.x * 256 + threadIdx.x;
    if (i < 384 * 128) {
        int n = i >> 7, k = i & 127;
        W2ht[i] = (_Float16)W2[k * 384 + n];
    }
}

// ---------------------------------------------------------------------------
// Kz: zero the accumulators (acc_s | acc_v contiguous)
// ---------------------------------------------------------------------------
__global__ void zero_kernel(float4* __restrict__ p, int n4) {
    int i = blockIdx.x * 256 + threadIdx.x;
    if (i < n4) p[i] = make_float4(0.f, 0.f, 0.f, 0.f);
}

// ---------------------------------------------------------------------------
// K1: node "pre" kernel. 16 nodes per 256-thread block (8 waves).
//   x_s = s @ W_pre_s / sqrt(128) + b_pre_s     (fp32 WMMA, K=128, N=128)
//   x_v = einsum(v, W_pre_v) / sqrt(64)         (fp32 WMMA per channel, K=64, N=64)
// ---------------------------------------------------------------------------
__global__ __launch_bounds__(256) void pre_kernel(
    const float* __restrict__ node_feat,
    const float* __restrict__ Wps, const float* __restrict__ bps,
    const float* __restrict__ Wpv,
    float* __restrict__ xs, float* __restrict__ xv)
{
    __shared__ __align__(16) float sh_node[16 * 320];
    const int tid = threadIdx.x;
    const int n0  = blockIdx.x * 16;

    for (int o = tid; o < 16 * 320; o += 256)
        sh_node[o] = node_feat[(size_t)n0 * 320 + o];
    __syncthreads();

    const int lane = tid & 31, wv = tid >> 5;
    const int m = lane & 15, hi = lane >> 4;

    // ---- x_s: wave wv owns N-tile wv (8 tiles x 16 cols = 128) ----
    {
        const int ncol = wv * 16 + m;
        v8f C = {0.f,0.f,0.f,0.f,0.f,0.f,0.f,0.f};
        for (int ks = 0; ks < 32; ++ks) {
            const int k = ks * 4 + 2 * hi;
            C = wmma_f32x4(sh_node[m * 320 + k], sh_node[m * 320 + k + 1],
                           Wps[k * 128 + ncol], Wps[(k + 1) * 128 + ncol], C);
        }
        const float bias = bps[ncol];
        #pragma unroll
        for (int r = 0; r < 8; ++r) {
            const int mr = r + 8 * hi;
            xs[(size_t)(n0 + mr) * 128 + ncol] = C[r] * 0.08838834764831845f + bias;
        }
    }

    // ---- x_v: 12 combos = 4 N-tiles x 3 channels, waves round-robin ----
    for (int combo = wv; combo < 12; combo += 8) {
        const int cch = combo % 3;
        const int nt  = combo / 3;
        const int ncol = nt * 16 + m;
        v8f C = {0.f,0.f,0.f,0.f,0.f,0.f,0.f,0.f};
        for (int ks = 0; ks < 16; ++ks) {
            const int k = ks * 4 + 2 * hi;
            C = wmma_f32x4(sh_node[m * 320 + 128 + k * 3 + cch],
                           sh_node[m * 320 + 128 + (k + 1) * 3 + cch],
                           Wpv[k * 64 + ncol], Wpv[(k + 1) * 64 + ncol], C);
        }
        #pragma unroll
        for (int r = 0; r < 8; ++r) {
            const int mr = r + 8 * hi;
            xv[(size_t)(n0 + mr) * 192 + ncol * 3 + cch] = C[r] * 0.125f;
        }
    }
}

// ---------------------------------------------------------------------------
// K2: fused edge kernel. 32 edges per 256-thread block (8 waves, 8000 blocks).
//   A) edge_attr staged via async global->LDS DMA (no VGPR round-trip),
//      then h = silu(edge_attr @ W1 + b1) (f32 VALU, tiny), stored f16 in LDS
//   B) w = h @ W2 + b2 via f16 WMMA: waves split 2 M-tiles x 24 N-tiles
//   C) per-edge messages + atomicAdd segment-sum (acc arrays live in L2)
// ---------------------------------------------------------------------------
__global__ __launch_bounds__(256) void edge_kernel(
    const float* __restrict__ edge_attr, const float* __restrict__ edge_rshs,
    const int* __restrict__ edge_index,
    const float* __restrict__ W1, const float* __restrict__ b1,
    const float* __restrict__ b2, const _Float16* __restrict__ W2ht,
    const float* __restrict__ xs, const float* __restrict__ xv,
    float* __restrict__ accs, float* __restrict__ accv)
{
    __shared__ __align__(16) float     sh_ea[32 * 16];   //  2 KB
    __shared__ __align__(16) _Float16  sh_h [32 * 128];  //  8 KB
    __shared__ __align__(16) float     sh_w [32 * 384];  // 48 KB

    const int tid = threadIdx.x;
    const int e0  = blockIdx.x * 32;

    // --- stage edge_attr: 2 KB via async global->LDS (ASYNCcnt), 8 B/thread ---
    {
        const unsigned int lds_dst = (unsigned int)(uintptr_t)&sh_ea[tid * 2];
        const unsigned long long gaddr =
            (unsigned long long)(const void*)(edge_attr + (size_t)e0 * 16 + tid * 2);
        asm volatile("global_load_async_to_lds_b64 %0, %1, off"
                     :: "v"(lds_dst), "v"(gaddr) : "memory");
    }
    // warm the scatter-phase operands while the DMA is in flight
    if (tid < 32) {
        __builtin_prefetch(edge_rshs + (size_t)(e0 + tid) * 4, 0, 3);
        __builtin_prefetch(edge_index + e0 + tid, 0, 3);
        __builtin_prefetch(edge_index + G_EDGES + e0 + tid, 0, 3);
    }
    asm volatile("s_wait_asynccnt 0x0" ::: "memory");
    __syncthreads();

    // --- phase A: hidden layer (16 -> 128), silu, store f16 ---
    for (int o = tid; o < 32 * 128; o += 256) {
        const int e = o >> 7, j = o & 127;
        float acc = b1[j];
        #pragma unroll
        for (int t = 0; t < 16; ++t)
            acc += sh_ea[e * 16 + t] * W1[t * 128 + j];
        sh_h[o] = (_Float16)(acc * fsigmoid(acc));
    }
    __syncthreads();

    // --- phase B: w = h @ W2 + b2 with f16 WMMA ---
    const int lane = tid & 31, wv = tid >> 5;
    const int m = lane & 15, hi = lane >> 4;
    const int ebase = (wv & 1) * 16;          // M-tile (16 edges)

    v16h A[4];
    #pragma unroll
    for (int kk = 0; kk < 4; ++kk) {          // K = 128 in 4 chunks of 32
        const int kc = kk * 32;
        union { v16h v; v8h h[2]; } u;
        u.h[0] = *reinterpret_cast<const v8h*>(&sh_h[(ebase + m) * 128 + kc + 8 * hi]);
        u.h[1] = *reinterpret_cast<const v8h*>(&sh_h[(ebase + m) * 128 + kc + 16 + 8 * hi]);
        A[kk] = u.v;
    }

    #pragma unroll
    for (int jj = 0; jj < 6; ++jj) {          // 4 wave-groups x 6 = 24 N-tiles
        const int n0c  = ((wv >> 1) * 6 + jj) * 16;
        const int ncol = n0c + m;
        v8f C = {0.f,0.f,0.f,0.f,0.f,0.f,0.f,0.f};
        #pragma unroll
        for (int kk = 0; kk < 4; ++kk) {
            const int kc = kk * 32;
            union { v16h v; v8h h[2]; } u;
            u.h[0] = *reinterpret_cast<const v8h*>(&W2ht[(size_t)ncol * 128 + kc + 8 * hi]);
            u.h[1] = *reinterpret_cast<const v8h*>(&W2ht[(size_t)ncol * 128 + kc + 16 + 8 * hi]);
            C = __builtin_amdgcn_wmma_f32_16x16x32_f16(false, A[kk], false, u.v,
                                                       (short)0, C, false, false);
        }
        const float bias = b2[ncol];
        #pragma unroll
        for (int r = 0; r < 8; ++r) {
            const int mr = r + 8 * hi;
            sh_w[(ebase + mr) * 384 + ncol] = C[r] + bias;
        }
    }
    __syncthreads();

    // --- phase C: messages + scatter. 8 threads per edge. ---
    const int eloc = tid >> 3, sub = tid & 7;
    const int eg   = e0 + eloc;
    const int dst  = edge_index[eg];
    const int src  = edge_index[G_EDGES + eg];
    const float r0  = edge_rshs[(size_t)eg * 4 + 0];
    const float r1x = edge_rshs[(size_t)eg * 4 + 1];
    const float r1y = edge_rshs[(size_t)eg * 4 + 2];
    const float r1z = edge_rshs[(size_t)eg * 4 + 3];
    const float* wrow = &sh_w[eloc * 384];
    const float* sj = &xs[(size_t)src * 128];
    const float* vj = &xv[(size_t)src * 192];
    float* as = &accs[(size_t)dst * 192];
    float* av = &accv[(size_t)dst * 576];

    for (int f = sub; f < 768; f += 8) {
        if (f < 128) {
            atomicAdd(&as[f], wrow[f] * sj[f] * r0);
        } else if (f < 192) {
            const int u = f - 128;
            const float dot3 = vj[u*3+0]*r1x + vj[u*3+1]*r1y + vj[u*3+2]*r1z;
            atomicAdd(&as[f], wrow[128 + u] * dot3 * 0.57735026918962576f);
        } else if (f < 576) {
            const int t = f - 192;
            const int i = t / 3, c = t - 3 * i;
            const float r1c = (c == 0) ? r1x : ((c == 1) ? r1y : r1z);
            atomicAdd(&av[i * 3 + c], wrow[192 + i] * sj[i] * r1c);
        } else {
            const int t = f - 576;
            const int u = t / 3, c = t - 3 * u;
            atomicAdd(&av[(128 + u) * 3 + c], wrow[320 + u] * vj[u * 3 + c] * r0);
        }
    }
}

// ---------------------------------------------------------------------------
// K3: node "post" kernel. 16 nodes per 256-thread block.
//   g_s = silu(acc_s); g_v = acc_v * sigmoid(|acc_v|)
//   p_s = g_s @ W_post_s / sqrt(192)   (fp32 WMMA, K=192, N=128)
//   p_v = einsum(g_v, W_post_v)/sqrt(192) (fp32 WMMA per channel, K=192, N=64)
//   out = concat(x_s + p_s, (x_v + p_v).reshape)
// ---------------------------------------------------------------------------
__global__ __launch_bounds__(256) void post_kernel(
    const float* __restrict__ accs, const float* __restrict__ accv,
    const float* __restrict__ Wposts, const float* __restrict__ Wpostv,
    const float* __restrict__ xs, const float* __restrict__ xv,
    float* __restrict__ out)
{
    __shared__ __align__(16) float sh_gs[16 * 192];  // 12 KB
    __shared__ __align__(16) float sh_gv[16 * 576];  // 36 KB
    const int tid = threadIdx.x;
    const int n0  = blockIdx.x * 16;

    for (int o = tid; o < 16 * 192; o += 256) {
        const float a = accs[(size_t)n0 * 192 + o];
        sh_gs[o] = a * fsigmoid(a);
    }
    for (int o = tid; o < 16 * 192; o += 256) {   // o = n_local*192 + channel
        const size_t base = (size_t)n0 * 576 + (size_t)o * 3;
        const float x = accv[base + 0];
        const float y = accv[base + 1];
        const float z = accv[base + 2];
        const float g = fsigmoid(sqrtf(x * x + y * y + z * z + 1e-12f));
        sh_gv[o * 3 + 0] = x * g;
        sh_gv[o * 3 + 1] = y * g;
        sh_gv[o * 3 + 2] = z * g;
    }
    __syncthreads();

    const int lane = tid & 31, wv = tid >> 5;
    const int m = lane & 15, hi = lane >> 4;
    const float inv192 = 0.07216878364870323f;

    // ---- p_s ----
    {
        const int ncol = wv * 16 + m;
        v8f C = {0.f,0.f,0.f,0.f,0.f,0.f,0.f,0.f};
        for (int ks = 0; ks < 48; ++ks) {
            const int k = ks * 4 + 2 * hi;
            C = wmma_f32x4(sh_gs[m * 192 + k], sh_gs[m * 192 + k + 1],
                           Wposts[k * 128 + ncol], Wposts[(k + 1) * 128 + ncol], C);
        }
        #pragma unroll
        for (int r = 0; r < 8; ++r) {
            const int mr = r + 8 * hi;
            out[(size_t)(n0 + mr) * 320 + ncol] =
                xs[(size_t)(n0 + mr) * 128 + ncol] + C[r] * inv192;
        }
    }

    // ---- p_v: 12 combos = 4 N-tiles x 3 channels ----
    for (int combo = wv; combo < 12; combo += 8) {
        const int cch = combo % 3;
        const int nt  = combo / 3;
        const int ncol = nt * 16 + m;
        v8f C = {0.f,0.f,0.f,0.f,0.f,0.f,0.f,0.f};
        for (int ks = 0; ks < 48; ++ks) {
            const int k = ks * 4 + 2 * hi;
            C = wmma_f32x4(sh_gv[m * 576 + k * 3 + cch],
                           sh_gv[m * 576 + (k + 1) * 3 + cch],
                           Wpostv[k * 64 + ncol], Wpostv[(k + 1) * 64 + ncol], C);
        }
        #pragma unroll
        for (int r = 0; r < 8; ++r) {
            const int mr = r + 8 * hi;
            out[(size_t)(n0 + mr) * 320 + 128 + ncol * 3 + cch] =
                xv[(size_t)(n0 + mr) * 192 + ncol * 3 + cch] + C[r] * inv192;
        }
    }
}

// ---------------------------------------------------------------------------
extern "C" void kernel_launch(void* const* d_in, const int* in_sizes, int n_in,
                              void* d_out, int out_size, void* d_ws, size_t ws_size,
                              hipStream_t stream)
{
    (void)in_sizes; (void)n_in; (void)out_size; (void)ws_size;

    const float* node_feat  = (const float*)d_in[0];
    const float* edge_attr  = (const float*)d_in[1];
    const float* edge_rshs  = (const float*)d_in[2];
    const int*   edge_index = (const int*)  d_in[3];
    const float* W_pre_s    = (const float*)d_in[4];
    const float* b_pre_s    = (const float*)d_in[5];
    const float* W_pre_v    = (const float*)d_in[6];
    const float* W1         = (const float*)d_in[7];
    const float* b1         = (const float*)d_in[8];
    const float* W2         = (const float*)d_in[9];
    const float* b2         = (const float*)d_in[10];
    const float* W_post_s   = (const float*)d_in[11];
    const float* W_post_v   = (const float*)d_in[12];
    float* out = (float*)d_out;

    // workspace layout (~66.5 MB): x_s | x_v | acc_s | acc_v | W2^T(f16)
    float* ws   = (float*)d_ws;
    float* xs   = ws;                                 // 16000*128
    float* xv   = xs   + (size_t)G_NODES * G_NS;      // 16000*192
    float* accs = xv   + (size_t)G_NODES * G_TPC;     // 16000*192
    float* accv = accs + (size_t)G_NODES * G_TPC;     // 16000*576
    _Float16* W2ht = (_Float16*)(accv + (size_t)G_NODES * G_TPC * 3);

    convert_w2_kernel<<<(384 * 128 + 255) / 256, 256, 0, stream>>>(W2, W2ht);

    const int n4 = G_NODES * (G_TPC + G_TPC * 3) / 4;   // acc_s + acc_v floats / 4
    zero_kernel<<<(n4 + 255) / 256, 256, 0, stream>>>((float4*)accs, n4);

    pre_kernel<<<G_NODES / 16, 256, 0, stream>>>(node_feat, W_pre_s, b_pre_s, W_pre_v, xs, xv);

    edge_kernel<<<G_EDGES / 32, 256, 0, stream>>>(edge_attr, edge_rshs, edge_index,
                                                  W1, b1, b2, W2ht, xs, xv, accs, accv);

    post_kernel<<<G_NODES / 16, 256, 0, stream>>>(accs, accv, W_post_s, W_post_v, xs, xv, out);
}
